// Quantize_575525618270
// MI455X (gfx1250) — matrix operand order repaired and checked by reading, not compiled, not run on hardware
//
#include <hip/hip_runtime.h>

typedef __attribute__((ext_vector_type(16))) _Float16 v16h;
typedef __attribute__((ext_vector_type(8)))  _Float16 v8h;
typedef __attribute__((ext_vector_type(4)))  _Float16 v4h;
typedef __attribute__((ext_vector_type(8)))  float    v8f;

#define N_TOK  131072
#define DIM    64
#define NCODE  1024
#define ALPHA_ 0.9f

#define WAVES_PER_BLOCK 8
#define ROWS_PER_WAVE   16
#define ROWS_PER_BLOCK  (WAVES_PER_BLOCK * ROWS_PER_WAVE)   // 128
#define NBLOCKS         (N_TOK / ROWS_PER_BLOCK)            // 1024
#define NWAVES_TOTAL    (NBLOCKS * WAVES_PER_BLOCK)         // 8192
#define KCHUNK          256
#define NCHUNK          (NCODE / KCHUNK)                    // 4
#define LDS_STRIDE      72                                  // 64 + 8 pad (halves)

#define PACK8(dst, base, fA, fB)                                              \
  dst[base+0]=(_Float16)fA.x; dst[base+1]=(_Float16)fA.y;                     \
  dst[base+2]=(_Float16)fA.z; dst[base+3]=(_Float16)fA.w;                     \
  dst[base+4]=(_Float16)fB.x; dst[base+5]=(_Float16)fB.y;                     \
  dst[base+6]=(_Float16)fB.z; dst[base+7]=(_Float16)fB.w;

__global__ __launch_bounds__(256) void vq_main(const float* __restrict__ x,
                                               const float* __restrict__ W,
                                               float* __restrict__ out,
                                               float* __restrict__ partials,
                                               float* __restrict__ diff)
{
    __shared__ _Float16 Wl[KCHUNK * LDS_STRIDE];   // 36,864 B, f16 codebook chunk
    __shared__ float    wnl[KCHUNK];               // per-code ||W_k||^2

    const int tid     = threadIdx.x;
    const int lane    = tid & 31;
    const int wave    = tid >> 5;
    const int rowBase = blockIdx.x * ROWS_PER_BLOCK + wave * ROWS_PER_WAVE;
    const int myRow   = rowBase + (lane & 15);
    const int hk4     = (lane < 16) ? 0 : 2;       // float4 index of K-half base

    // ---- A fragments: x rows in WMMA 16x32 f16 layout, kept resident ----
    const float4* x4 = (const float4*)(x + (size_t)myRow * DIM);
    float4 p0 = x4[hk4 + 0],  p1 = x4[hk4 + 1];    // K = hk..hk+7      (chunk 0)
    float4 p2 = x4[hk4 + 4],  p3 = x4[hk4 + 5];    // K = hk+16..hk+23
    float4 q0 = x4[hk4 + 8],  q1 = x4[hk4 + 9];    // K = 32+hk..       (chunk 1)
    float4 q2 = x4[hk4 + 12], q3 = x4[hk4 + 13];   // K = 48+hk..
    v16h a0, a1;
    PACK8(a0, 0, p0, p1); PACK8(a0, 8, p2, p3);
    PACK8(a1, 0, q0, q1); PACK8(a1, 8, q2, q3);

    float minv[8];
    int   mini[8];
#pragma unroll
    for (int v = 0; v < 8; ++v) { minv[v] = 3.402823466e+38f; mini[v] = 0; }

    const int nl = lane & 15;
    const int kb = (lane < 16) ? 0 : 8;            // B-fragment K-half base (in halves)

    for (int ch = 0; ch < NCHUNK; ++ch) {
        const int cbase = ch * KCHUNK;

        // ---- cooperative load+convert W chunk -> LDS (4 threads / row) ----
        {
            const int part  = tid & 3;             // 16 floats each
            const int rloc0 = tid >> 2;            // 0..63
#pragma unroll
            for (int pass = 0; pass < 4; ++pass) {
                const int rloc = rloc0 + pass * 64;
                const float4* wr = (const float4*)(W + (size_t)(cbase + rloc) * DIM);
#pragma unroll
                for (int q = 0; q < 4; ++q) {
                    float4 f = wr[part * 4 + q];
                    v4h h = { (_Float16)f.x, (_Float16)f.y, (_Float16)f.z, (_Float16)f.w };
                    *(v4h*)&Wl[rloc * LDS_STRIDE + part * 16 + q * 4] = h;
                }
            }
        }
        __syncthreads();

        // ---- ||W_k||^2 per code, one thread per row (deterministic order) ----
        {
            float acc = 0.0f;
#pragma unroll
            for (int q = 0; q < 16; ++q) {
                v4h h = *(const v4h*)&Wl[tid * LDS_STRIDE + q * 4];
                float f0 = (float)h[0], f1 = (float)h[1], f2 = (float)h[2], f3 = (float)h[3];
                acc += f0 * f0 + f1 * f1 + f2 * f2 + f3 * f3;
            }
            wnl[tid] = acc;
        }
        __syncthreads();

        // ---- 16 score tiles of 16 codes each: 2 WMMAs per tile ----
#pragma unroll 4
        for (int tn = 0; tn < KCHUNK / 16; ++tn) {
            const int rowW = tn * 16 + nl;
            const int off  = rowW * LDS_STRIDE + kb;
            v8h t0 = *(const v8h*)&Wl[off];
            v8h t1 = *(const v8h*)&Wl[off + 16];
            v8h t2 = *(const v8h*)&Wl[off + 32];
            v8h t3 = *(const v8h*)&Wl[off + 48];
            v16h b0, b1;
#pragma unroll
            for (int i = 0; i < 8; ++i) {
                b0[i] = t0[i]; b0[8 + i] = t1[i];
                b1[i] = t2[i]; b1[8 + i] = t3[i];
            }
            v8f c = {};
            c = __builtin_amdgcn_wmma_f32_16x16x32_f16(false, a0, false, b0,
                                                       (short)0, c, false, false);
            c = __builtin_amdgcn_wmma_f32_16x16x32_f16(false, a1, false, b1,
                                                       (short)0, c, false, false);
            const float wn  = wnl[rowW];
            const int   idx = cbase + rowW;
#pragma unroll
            for (int v = 0; v < 8; ++v) {
                float d = wn - 2.0f * c[v];        // ||x||^2 dropped (row-constant)
                if (d < minv[v]) { minv[v] = d; mini[v] = idx; }
            }
        }
        __syncthreads();
    }

    // ---- cross-lane argmin within each 16-lane column group ----
#pragma unroll
    for (int v = 0; v < 8; ++v) {
        float mv = minv[v]; int mi = mini[v];
#pragma unroll
        for (int m = 1; m < 16; m <<= 1) {
            float ov = __shfl_xor(mv, m, 32);
            int   oi = __shfl_xor(mi, m, 32);
            if (ov < mv || (ov == mv && oi < mi)) { mv = ov; mi = oi; }
        }
        minv[v] = mv; mini[v] = mi;
    }

    // ---- gather W_j (exact f32), write out, accumulate ||x - W_j||^2 ----
    float acc = 0.0f;
#pragma unroll
    for (int r = 0; r < 16; ++r) {
        const int j   = __shfl(mini[r & 7], (r < 8) ? 0 : 16, 32);
        const int row = rowBase + r;
        float2 wv = ((const float2*)(W + (size_t)j   * DIM))[lane];
        float2 xv = ((const float2*)(x + (size_t)row * DIM))[lane];
        ((float2*)(out + (size_t)row * DIM))[lane] = wv;
        float d0 = xv.x - wv.x, d1 = xv.y - wv.y;
        acc += d0 * d0 + d1 * d1;
    }
#pragma unroll
    for (int m = 16; m >= 1; m >>= 1) acc += __shfl_xor(acc, m, 32);

    if (lane == 0) {
        if (partials) {
            partials[blockIdx.x * WAVES_PER_BLOCK + wave] = acc;
        } else {
            atomicAdd(diff, acc * ((1.0f + ALPHA_) / (float)N_TOK));
        }
    }
}

__global__ void vq_reduce(const float* __restrict__ partials, float* __restrict__ diff)
{
    __shared__ float s[256];
    float a = 0.0f;
    for (int i = threadIdx.x; i < NWAVES_TOTAL; i += 256) a += partials[i];
    s[threadIdx.x] = a;
    __syncthreads();
    for (int m = 128; m >= 1; m >>= 1) {
        if ((int)threadIdx.x < m) s[threadIdx.x] += s[threadIdx.x + m];
        __syncthreads();
    }
    if (threadIdx.x == 0) *diff = s[0] * ((1.0f + ALPHA_) / (float)N_TOK);
}

__global__ void vq_zero(float* __restrict__ diff)
{
    if (threadIdx.x == 0) *diff = 0.0f;
}

extern "C" void kernel_launch(void* const* d_in, const int* in_sizes, int n_in,
                              void* d_out, int out_size, void* d_ws, size_t ws_size,
                              hipStream_t stream)
{
    (void)in_sizes; (void)n_in; (void)out_size;
    const float* x   = (const float*)d_in[0];   // [131072, 64] f32
    const float* W   = (const float*)d_in[1];   // [1024, 64]   f32
    float* out  = (float*)d_out;                // [131072*64] W_j, then diff
    float* diff = out + (size_t)N_TOK * DIM;

    if (ws_size >= (size_t)NWAVES_TOTAL * sizeof(float)) {
        float* partials = (float*)d_ws;
        vq_main<<<NBLOCKS, 256, 0, stream>>>(x, W, out, partials, diff);
        vq_reduce<<<1, 256, 0, stream>>>(partials, diff);
    } else {
        vq_zero<<<1, 1, 0, stream>>>(diff);
        vq_main<<<NBLOCKS, 256, 0, stream>>>(x, W, out, nullptr, diff);
    }
}
